// PrototypeUpdater_74723841015972
// MI455X (gfx1250) — matrix operand order repaired and checked by reading, not compiled, not run on hardware
//
#include <hip/hip_runtime.h>
#include <hip/hip_bf16.h>
#include <math.h>

// ---------------------------------------------------------------------------
// MI455X (gfx1250, wave32) implementation.
// Dominant cost: [65536,512]x[512,256] GEMM -> bf16 WMMA 16x16x32, f32 accum.
// ---------------------------------------------------------------------------

typedef __attribute__((ext_vector_type(16))) __bf16 v16bf;
typedef __attribute__((ext_vector_type(8)))  __bf16 v8bf;
typedef __attribute__((ext_vector_type(8)))  float  v8f;

#define BB   16
#define LL   4096
#define DD   256
#define KDIM 512              // 2*D
#define RHO_C 0.5f
#define LQ_C  1.0f
#define LS_C  1.0f
#define SCALE_C 0.0625f       // 1/sqrt(256)

#define ROWS_PER_GROUP 64
#define ROWGROUPS      8
#define ROWS_PER_BLOCK (ROWS_PER_GROUP * ROWGROUPS)   // 512
#define COLS_PER_BLOCK 128                            // gridDim.y = 2 halves
#define WSTRIDE (KDIM + 8)                            // bf16 elems, pad vs banks

__device__ __forceinline__ void pack8(v16bf& a, int base, float4 u, float4 v) {
  a[base + 0] = (__bf16)u.x; a[base + 1] = (__bf16)u.y;
  a[base + 2] = (__bf16)u.z; a[base + 3] = (__bf16)u.w;
  a[base + 4] = (__bf16)v.x; a[base + 5] = (__bf16)v.y;
  a[base + 6] = (__bf16)v.z; a[base + 7] = (__bf16)v.w;
}

__device__ __forceinline__ float4 f4sub(float4 a, float4 b) {
  return make_float4(a.x - b.x, a.y - b.y, a.z - b.z, a.w - b.w);
}

// ---------------------------------------------------------------------------
// Pass 1: o = tanh([h | rp-rn] @ W^T + b)  (bf16 WMMA),
//         fused per-row logit partials  o . (a + rho*q[b]).
// Block: 256 thr = 8 waves; covers 512 rows x 128 cols (8 row-groups of 64).
// Wave w: row-tile (w&3)*16, col-group (w>>2)*64 -> 4 accum tiles of 16x16.
// W (128 cols x 512 K) staged once per block into LDS as bf16.
// K-loop split into a straight-line h half and a straight-line (rp-rn) half.
// ---------------------------------------------------------------------------
__global__ __launch_bounds__(256)
void pu_pass1_wmma(const float* __restrict__ h,  const float* __restrict__ rp,
                   const float* __restrict__ rn, const float* __restrict__ q,
                   const float* __restrict__ prototypes,
                   const int*   __restrict__ proto_idx,
                   const float* __restrict__ Ww, const float* __restrict__ Wb,
                   __bf16* __restrict__ o_out, float* __restrict__ logits_part)
{
  __shared__ __align__(16) __bf16 wlds[COLS_PER_BLOCK * WSTRIDE];

  const int tid     = threadIdx.x;
  const int ch      = blockIdx.y;                 // column half (0/1)
  const int nbase_g = ch * COLS_PER_BLOCK;        // global col base of block

  // ---- stage W[nbase_g .. +128) x [0..512) fp32 -> bf16 into LDS ----------
  for (int i = 0; i < 64; ++i) {
    const int idx4 = tid + i * 256;               // 16384 float4s total
    const int n    = idx4 >> 7;                   // 128 float4 per row
    const int k4   = (idx4 & 127) << 2;
    const float4 w4 = *(const float4*)(Ww + (size_t)(nbase_g + n) * KDIM + k4);
    __bf16* dst = &wlds[n * WSTRIDE + k4];
    dst[0] = (__bf16)w4.x; dst[1] = (__bf16)w4.y;
    dst[2] = (__bf16)w4.z; dst[3] = (__bf16)w4.w;
  }
  __syncthreads();

  const int lane    = tid & 31;
  const int wv      = tid >> 5;
  const int mtile   = wv & 3;                     // 0..3 (16-row tile)
  const int cg      = wv >> 2;                    // 0/1  (64-col group)
  const int nbase_w = cg * 64;
  const int laneM   = lane & 15;
  const int laneHi  = lane >> 4;                  // 0/1

  const int b    = blockIdx.x >> 3;               // 512 rows/block, 4096 rows/b
  const int pidx = proto_idx[0];

  // per-lane column-dependent constants (cols fixed for whole block)
  float wvt[4], bvt[4];
  int   colg[4];
  #pragma unroll
  for (int t = 0; t < 4; ++t) {
    const int c = nbase_g + nbase_w + t * 16 + laneM;
    colg[t] = c;
    wvt[t]  = prototypes[pidx * DD + c] + RHO_C * q[b * DD + c];
    bvt[t]  = Wb[c];
  }

  for (int rg = 0; rg < ROWGROUPS; ++rg) {
    const int rbase = blockIdx.x * ROWS_PER_BLOCK + rg * ROWS_PER_GROUP
                    + mtile * 16;
    const int rowA  = rbase + laneM;              // row this lane fetches for A
    const float* hrow  = h  + (size_t)rowA * DD;
    const float* rprow = rp + (size_t)rowA * DD;
    const float* rnrow = rn + (size_t)rowA * DD;

    // prefetch this group's rp/rn rows (consumed in the second k-half)
    #pragma unroll
    for (int j = 0; j < 8; ++j) {
      __builtin_prefetch(rprow + j * 32, 0, 1);   // -> global_prefetch_b8
      __builtin_prefetch(rnrow + j * 32, 0, 1);
    }

    v8f acc[4] = {};

    // ---- K half 1: inp = h, k in [0,256) -------------------------------
    for (int ks = 0; ks < 8; ++ks) {
      const int k0 = ks * 32;
      const int kr = k0 + laneHi * 8;   // A: lanes<16 K{0-7,16-23}, hi K{8-15,24-31}
      const float* p = hrow + kr;
      v16bf afrag;
      pack8(afrag, 0, *(const float4*)(p),      *(const float4*)(p + 4));
      pack8(afrag, 8, *(const float4*)(p + 16), *(const float4*)(p + 20));

      const int kb = k0 + laneHi * 8;   // B: lanes<16 K{0-7,16-23}, hi K{8-15,24-31}
      #pragma unroll
      for (int t = 0; t < 4; ++t) {
        const __bf16* src = &wlds[(nbase_w + t * 16 + laneM) * WSTRIDE + kb];
        v16bf bfrag;
        *((v8bf*)&bfrag)     = *(const v8bf*)(src);
        *((v8bf*)&bfrag + 1) = *(const v8bf*)(src + 16);
        acc[t] = __builtin_amdgcn_wmma_f32_16x16x32_bf16(
            false, afrag, false, bfrag, (short)0, acc[t], false, false);
      }
    }

    // prefetch next group's h rows while the rp-rn half runs
    if (rg + 1 < ROWGROUPS) {
      const float* nh = hrow + (size_t)ROWS_PER_GROUP * DD;
      #pragma unroll
      for (int j = 0; j < 8; ++j) {
        __builtin_prefetch(nh + j * 32, 0, 1);
      }
    }

    // ---- K half 2: inp = rp - rn, k in [256,512) -----------------------
    for (int ks = 0; ks < 8; ++ks) {
      const int k0 = ks * 32;
      const int kr = k0 + laneHi * 8;
      const float* pp = rprow + kr;
      const float* pn = rnrow + kr;
      const float4 u0 = f4sub(*(const float4*)(pp),      *(const float4*)(pn));
      const float4 u1 = f4sub(*(const float4*)(pp + 4),  *(const float4*)(pn + 4));
      const float4 u2 = f4sub(*(const float4*)(pp + 16), *(const float4*)(pn + 16));
      const float4 u3 = f4sub(*(const float4*)(pp + 20), *(const float4*)(pn + 20));
      v16bf afrag;
      pack8(afrag, 0, u0, u1);
      pack8(afrag, 8, u2, u3);

      const int kb = 256 + k0 + laneHi * 8;
      #pragma unroll
      for (int t = 0; t < 4; ++t) {
        const __bf16* src = &wlds[(nbase_w + t * 16 + laneM) * WSTRIDE + kb];
        v16bf bfrag;
        *((v8bf*)&bfrag)     = *(const v8bf*)(src);
        *((v8bf*)&bfrag + 1) = *(const v8bf*)(src + 16);
        acc[t] = __builtin_amdgcn_wmma_f32_16x16x32_bf16(
            false, afrag, false, bfrag, (short)0, acc[t], false, false);
      }
    }

    // epilogue: bias + tanh, store o (bf16), per-row logit partials
    float part[8] = {0.f, 0.f, 0.f, 0.f, 0.f, 0.f, 0.f, 0.f};
    #pragma unroll
    for (int t = 0; t < 4; ++t) {
      #pragma unroll
      for (int j = 0; j < 8; ++j) {
        const float ov = tanhf(acc[t][j] + bvt[t]);
        const int row  = rbase + j + laneHi * 8;  // C layout: vgpr j -> M=j / j+8
        o_out[(size_t)row * DD + colg[t]] = (__bf16)ov;
        part[j] += ov * wvt[t];
      }
    }
    // reduce over the 16 lanes of each half (cols), keep per-row sums
    #pragma unroll
    for (int j = 0; j < 8; ++j) {
      float v = part[j];
      v += __shfl_xor(v, 1);
      v += __shfl_xor(v, 2);
      v += __shfl_xor(v, 4);
      v += __shfl_xor(v, 8);
      part[j] = v;
    }
    if (laneM == 0) {
      const int pI = ch * 2 + cg;                 // 4 deterministic partials/row
      #pragma unroll
      for (int j = 0; j < 8; ++j) {
        const int row = rbase + j + laneHi * 8;
        logits_part[(size_t)row * 4 + pI] = part[j];
      }
    }
  }
}

// ---------------------------------------------------------------------------
// Pass 2: per-batch masked softmax over L, then s_q[b,d] = sum_l alpha*o.
// ---------------------------------------------------------------------------
__global__ __launch_bounds__(256)
void pu_pass2_softmax(const float* __restrict__ logits_part,
                      const __bf16* __restrict__ o_in,
                      const int* __restrict__ lengths,
                      float* __restrict__ s_q)
{
  __shared__ float llds[LL];
  __shared__ float red[8];
  const int b   = blockIdx.x;
  const int tid = threadIdx.x;
  const int len = lengths[b];

  float lmax = -3.0e38f;
  for (int i = 0; i < LL / 256; ++i) {
    const int l = tid + i * 256;
    const float4 lp = ((const float4*)logits_part)[(size_t)b * LL + l];
    float s = (lp.x + lp.y + lp.z + lp.w) * SCALE_C;
    if (l >= len) s = -1.0e30f;
    llds[l] = s;
    lmax = fmaxf(lmax, s);
  }
  for (int m = 16; m >= 1; m >>= 1) lmax = fmaxf(lmax, __shfl_xor(lmax, m));
  if ((tid & 31) == 0) red[tid >> 5] = lmax;
  __syncthreads();
  float gmax = red[0];
  for (int i = 1; i < 8; ++i) gmax = fmaxf(gmax, red[i]);

  float lsum = 0.f;
  for (int i = 0; i < LL / 256; ++i) {
    const int l = tid + i * 256;
    const float e = __expf(llds[l] - gmax);
    llds[l] = e;
    lsum += e;
  }
  for (int m = 16; m >= 1; m >>= 1) lsum += __shfl_xor(lsum, m);
  __syncthreads();
  if ((tid & 31) == 0) red[tid >> 5] = lsum;
  __syncthreads();
  float gsum = 0.f;
  for (int i = 0; i < 8; ++i) gsum += red[i];
  const float inv = 1.0f / gsum;

  // s_q: thread owns column d = tid; coalesced 512B bf16 row reads
  float acc = 0.f;
  const __bf16* ob = o_in + (size_t)b * LL * DD + tid;
  #pragma unroll 8
  for (int l = 0; l < LL; ++l) {
    acc += llds[l] * (float)ob[(size_t)l * DD];
  }
  s_q[b * DD + tid] = acc * inv;
}

// ---------------------------------------------------------------------------
// Pass 3: z = a + Lq*(q@Wq^T+bq) + Ls*(s_q@Ws^T+bs); delta; mean softplus(-d).
// ---------------------------------------------------------------------------
__global__ __launch_bounds__(256)
void pu_pass3_loss(const float* __restrict__ q,   const float* __restrict__ r_pos,
                   const float* __restrict__ r_neg,
                   const float* __restrict__ prototypes,
                   const int*   __restrict__ proto_idx,
                   const float* __restrict__ Wq_w, const float* __restrict__ Wq_b,
                   const float* __restrict__ Ws_w, const float* __restrict__ Ws_b,
                   const float* __restrict__ s_q,  float* __restrict__ out)
{
  __shared__ float red[8];
  __shared__ float dsh[BB];
  const int d    = threadIdx.x;
  const int pidx = proto_idx[0];
  const float a_d = prototypes[pidx * DD + d];
  const float* Wqr = Wq_w + d * DD;
  const float* Wsr = Ws_w + d * DD;

  for (int b = 0; b < BB; ++b) {
    const float* qb = q   + b * DD;
    const float* sb = s_q + b * DD;
    float zq = 0.f, zs = 0.f;
    for (int e = 0; e < DD; ++e) {
      zq = fmaf(qb[e], Wqr[e], zq);
      zs = fmaf(sb[e], Wsr[e], zs);
    }
    const float z  = a_d + LQ_C * (zq + Wq_b[d]) + LS_C * (zs + Ws_b[d]);
    float v = z * (r_pos[b * DD + d] - r_neg[b * DD + d]);
    for (int m = 16; m >= 1; m >>= 1) v += __shfl_xor(v, m);
    if ((d & 31) == 0) red[d >> 5] = v;
    __syncthreads();
    if (d == 0) {
      float delta = 0.f;
      for (int i = 0; i < 8; ++i) delta += red[i];
      const float x = -delta;
      dsh[b] = fmaxf(x, 0.f) + log1pf(__expf(-fabsf(x)));   // softplus(-delta)
    }
    __syncthreads();
  }
  if (d == 0) {
    float m = 0.f;
    for (int b = 0; b < BB; ++b) m += dsh[b];
    out[0] = m / (float)BB;
  }
}

// ---------------------------------------------------------------------------
extern "C" void kernel_launch(void* const* d_in, const int* in_sizes, int n_in,
                              void* d_out, int out_size, void* d_ws, size_t ws_size,
                              hipStream_t stream)
{
  const float* q    = (const float*)d_in[0];
  const float* rpos = (const float*)d_in[1];
  const float* rneg = (const float*)d_in[2];
  const float* h    = (const float*)d_in[3];
  const float* rp   = (const float*)d_in[4];
  const float* rn   = (const float*)d_in[5];
  const int*   len  = (const int*)d_in[6];
  const int*   pix  = (const int*)d_in[7];
  const float* prot = (const float*)d_in[8];
  const float* Ww   = (const float*)d_in[9];
  const float* Wb   = (const float*)d_in[10];
  const float* Wqw  = (const float*)d_in[11];
  const float* Wqb  = (const float*)d_in[12];
  const float* Wsw  = (const float*)d_in[13];
  const float* Wsb  = (const float*)d_in[14];

  // workspace layout: o (bf16, 32MB) | logits_part (f32, 1MB) | s_q (f32, 16KB)
  __bf16* o_ws = (__bf16*)d_ws;
  float*  logits_part = (float*)((char*)d_ws + (size_t)BB * LL * DD * sizeof(__bf16));
  float*  s_q = (float*)((char*)logits_part + (size_t)BB * LL * 4 * sizeof(float));

  dim3 g1(BB * LL / ROWS_PER_BLOCK, 2);   // 128 x 2
  pu_pass1_wmma<<<g1, 256, 0, stream>>>(h, rp, rn, q, prot, pix, Ww, Wb,
                                        o_ws, logits_part);
  pu_pass2_softmax<<<BB, 256, 0, stream>>>(logits_part, o_ws, len, s_q);
  pu_pass3_loss<<<1, 256, 0, stream>>>(q, rpos, rneg, prot, pix,
                                       Wqw, Wqb, Wsw, Wsb, s_q, (float*)d_out);
}
